// MultiHeadAttention_34840774705440
// MI455X (gfx1250) — matrix operand order repaired and checked by reading, not compiled, not run on hardware
//
#include <hip/hip_runtime.h>
#include <hip/hip_bf16.h>

// ---------------------------------------------------------------------------
// MultiHeadAttention (B=2, S=2048, D=1024, H=16, Hd=64), causal, fp32 in/out.
// Pipeline: fp32->bf16 convert, 3x WMMA GEMM (Q,K,V), flash-attention kernel,
// final WMMA GEMM with bias -> fp32 out.
// CDNA5 paths: v_wmma_f32_16x16x32_bf16, global_load_async_to_lds_b128 (+
// s_wait_asynccnt) double-buffered staging.
// ---------------------------------------------------------------------------

#define EMB    1024
#define NHEADS 16
#define HD     64
#define BATCH  2
#define SEQ    2048
#define MTOT   (BATCH * SEQ)          // 4096 tokens total

#define NEG_INF (-__builtin_huge_valf())
#define LOG2E   1.44269504088896340736f

typedef __attribute__((ext_vector_type(16))) __bf16 v16bf;
typedef __attribute__((ext_vector_type(8)))  float  v8f;
typedef int v4i __attribute__((vector_size(16)));

union Frag {
    v16bf v;
    uint4 q[2];
};

__device__ __forceinline__ unsigned short f32_to_bf16_rn(float f) {
    unsigned int u = __float_as_uint(f);
    unsigned int r = u + 0x7FFFu + ((u >> 16) & 1u);   // round to nearest even
    return (unsigned short)(r >> 16);
}

// ---------------------------------------------------------------------------
// Async global->LDS copy (16B per lane). Uses the gfx1250 ASYNCcnt path when
// the toolchain declares the builtin; otherwise a plain VGPR-bounce copy.
// ---------------------------------------------------------------------------
#if defined(__has_builtin)
#if __has_builtin(__builtin_amdgcn_global_load_async_to_lds_b128)
#define HAVE_ASYNC_LDS 1
#endif
#endif

__device__ __forceinline__ void async_b128(const void* gptr, void* lptr) {
#ifdef HAVE_ASYNC_LDS
    __builtin_amdgcn_global_load_async_to_lds_b128(
        (__attribute__((address_space(1))) v4i*)gptr,
        (__attribute__((address_space(3))) v4i*)lptr, 0, 0);
#else
    *(uint4*)lptr = *(const uint4*)gptr;
#endif
}

__device__ __forceinline__ void wait_async_all() {
#ifdef HAVE_ASYNC_LDS
#if __has_builtin(__builtin_amdgcn_s_wait_asynccnt)
    __builtin_amdgcn_s_wait_asynccnt(0);
#else
    asm volatile("s_wait_asynccnt 0x0" ::: "memory");
#endif
#endif
}

// ---------------------------------------------------------------------------
// fp32 -> bf16 conversion (vectorized x4)
// ---------------------------------------------------------------------------
__global__ void cvt_f32_bf16(const float4* __restrict__ in,
                             ushort4* __restrict__ out, int n4) {
    int i = blockIdx.x * blockDim.x + threadIdx.x;
    if (i < n4) {
        float4 f = in[i];
        ushort4 o;
        o.x = f32_to_bf16_rn(f.x);
        o.y = f32_to_bf16_rn(f.y);
        o.z = f32_to_bf16_rn(f.z);
        o.w = f32_to_bf16_rn(f.w);
        out[i] = o;
    }
}

// ---------------------------------------------------------------------------
// C[M,N] = A[M,K] @ B[N,K]^T   (both bf16 row-major, K contiguous => NT gemm)
// 256 threads = 8 waves; block tile 128x128; wave tile 32x64 (2x4 WMMA tiles).
// Double-buffered async LDS staging: issue slab i+1, compute slab i.
// Optional bf16 output (for Q/K/V) and/or f32 output with bias (final proj).
// ---------------------------------------------------------------------------
#define GT_M 128
#define GT_N 128
#define GT_K 64
#define GLDK 72   // padded LDS row stride in halfs (144B = conflict-free b128)

__global__ __launch_bounds__(256, 1)
void gemm_bf16_nt(const unsigned short* __restrict__ A,
                  const unsigned short* __restrict__ Bm,
                  unsigned short* __restrict__ Cbf,
                  float* __restrict__ Cf,
                  const float* __restrict__ bias,
                  int M, int N, int K) {
    __shared__ unsigned short As[2][GT_M][GLDK];
    __shared__ unsigned short Bs[2][GT_N][GLDK];

    const int tid  = threadIdx.x;
    const int lane = tid & 31;
    const int wid  = tid >> 5;     // 0..7
    const int wm   = wid & 3;      // 4 waves along M
    const int wn   = wid >> 2;     // 2 waves along N
    const int lm   = lane & 15;
    const int g    = lane >> 4;

    const int mBase = blockIdx.y * GT_M;
    const int nBase = blockIdx.x * GT_N;

    // Stage one 128x64-half slab of A and B into LDS buffer `buf`.
    auto stage = [&](int buf, int k0) {
#pragma unroll
        for (int c = 0; c < 4; c++) {
            int chunk = tid + c * 256;   // 0..1023 16B-chunks
            int row   = chunk >> 3;
            int col   = (chunk & 7) * 8;
            async_b128(A  + (size_t)(mBase + row) * K + k0 + col, &As[buf][row][col]);
            async_b128(Bm + (size_t)(nBase + row) * K + k0 + col, &Bs[buf][row][col]);
        }
    };

    const v8f vzero = {0.f, 0.f, 0.f, 0.f, 0.f, 0.f, 0.f, 0.f};
    v8f acc[2][4];
#pragma unroll
    for (int mi = 0; mi < 2; mi++)
#pragma unroll
        for (int ni = 0; ni < 4; ni++) acc[mi][ni] = vzero;

    const int nk = K / GT_K;
    stage(0, 0);

    for (int it = 0; it < nk; it++) {
        wait_async_all();      // slab `it` resident in LDS (this wave's part)
        __syncthreads();       // all waves' parts resident; prev compute done
        if (it + 1 < nk) stage((it + 1) & 1, (it + 1) * GT_K);

        const int buf = it & 1;
#pragma unroll
        for (int ks = 0; ks < GT_K; ks += 32) {
            Frag a[2], bfr[4];
#pragma unroll
            for (int mi = 0; mi < 2; mi++) {
                const unsigned short* p = &As[buf][wm * 32 + mi * 16 + lm][ks];
                a[mi].q[0] = *(const uint4*)(p + 8 * g);        // K = 8g..8g+7
                a[mi].q[1] = *(const uint4*)(p + 16 + 8 * g);   // K = 16+8g..
            }
#pragma unroll
            for (int ni = 0; ni < 4; ni++) {
                const unsigned short* p = &Bs[buf][wn * 64 + ni * 16 + lm][ks];
                bfr[ni].q[0] = *(const uint4*)(p + 8 * g);      // K = 8g..8g+7
                bfr[ni].q[1] = *(const uint4*)(p + 16 + 8 * g); // K = 16+8g..
            }
#pragma unroll
            for (int mi = 0; mi < 2; mi++)
#pragma unroll
                for (int ni = 0; ni < 4; ni++)
                    acc[mi][ni] = __builtin_amdgcn_wmma_f32_16x16x32_bf16(
                        false, a[mi].v, false, bfr[ni].v,
                        (short)0, acc[mi][ni], false, false);
        }
        __syncthreads();       // compute done before next slab overwrites buf
    }

    // C layout: element (M = r + 8g, N = lm) in VGPR r.
    if (Cf) {
#pragma unroll
        for (int mi = 0; mi < 2; mi++)
#pragma unroll
            for (int ni = 0; ni < 4; ni++)
#pragma unroll
                for (int r = 0; r < 8; r++) {
                    int m = mBase + wm * 32 + mi * 16 + r + 8 * g;
                    int n = nBase + wn * 64 + ni * 16 + lm;
                    Cf[(size_t)m * N + n] =
                        acc[mi][ni][r] + (bias ? bias[n] : 0.0f);
                }
    } else {
#pragma unroll
        for (int mi = 0; mi < 2; mi++)
#pragma unroll
            for (int ni = 0; ni < 4; ni++)
#pragma unroll
                for (int r = 0; r < 8; r++) {
                    int m = mBase + wm * 32 + mi * 16 + r + 8 * g;
                    int n = nBase + wn * 64 + ni * 16 + lm;
                    Cbf[(size_t)m * N + n] = f32_to_bf16_rn(acc[mi][ni][r]);
                }
    }
}

// ---------------------------------------------------------------------------
// Causal flash attention. Grid: (S/128, H, B), block = 128 threads (4 waves).
// Each wave owns 32 query rows; stream 64-key tiles with online softmax.
// Q/K tiles staged with async-to-LDS; V staged with a transpose on store.
// Scale = 1/sqrt(64) applied after masking (== reference order).
// ---------------------------------------------------------------------------
#define AQ  128
#define AK  64
#define ALD 72

__global__ __launch_bounds__(128, 1)
void attn_causal(const unsigned short* __restrict__ Qh,
                 const unsigned short* __restrict__ Kh,
                 const unsigned short* __restrict__ Vh,
                 unsigned short* __restrict__ Ctx) {
    __shared__ unsigned short Qs[AQ][ALD];   // queries, row-major (hd contig)
    __shared__ unsigned short Ks[AK][ALD];   // keys,    row-major (hd contig)
    __shared__ unsigned short Vt[AK][ALD];   // V^T: [hd][key] (key contig)
    __shared__ unsigned short Ps[AQ][ALD];   // softmax probs, bf16

    const int tid  = threadIdx.x;
    const int lane = tid & 31;
    const int wid  = tid >> 5;   // 0..3
    const int lm   = lane & 15;
    const int g    = lane >> 4;

    const int q0 = blockIdx.x * AQ;
    const int h  = blockIdx.y;
    const int b  = blockIdx.z;

    const size_t headOff = (size_t)h * HD;
    const size_t batOff  = (size_t)b * SEQ * EMB;

    // Stage Q tile (async): 128 rows x 64 halfs = 1024 chunks, 8 per thread.
#pragma unroll
    for (int c = 0; c < 8; c++) {
        int chunk = tid + c * 128;
        int row   = chunk >> 3;
        int col   = (chunk & 7) * 8;
        async_b128(Qh + batOff + (size_t)(q0 + row) * EMB + headOff + col,
                   &Qs[row][col]);
    }

    const v8f vzero = {0.f, 0.f, 0.f, 0.f, 0.f, 0.f, 0.f, 0.f};
    v8f o[2][4];
#pragma unroll
    for (int mi = 0; mi < 2; mi++)
#pragma unroll
        for (int ni = 0; ni < 4; ni++) o[mi][ni] = vzero;

    float mrun[2][8], lrun[2][8];
#pragma unroll
    for (int mi = 0; mi < 2; mi++)
#pragma unroll
        for (int r = 0; r < 8; r++) { mrun[mi][r] = NEG_INF; lrun[mi][r] = 0.f; }

    const int nkt = (q0 + AQ) / AK;   // causal: only tiles with keys <= q0+127
    for (int kt = 0; kt < nkt; kt++) {
        const int kb = kt * AK;

        // Stage K (async, row-major) and V (sync, transposed): 512 chunks.
#pragma unroll
        for (int c = 0; c < 4; c++) {
            int chunk = tid + c * 128;
            int row   = chunk >> 3;
            int col   = (chunk & 7) * 8;
            async_b128(Kh + batOff + (size_t)(kb + row) * EMB + headOff + col,
                       &Ks[row][col]);
            uint4 vq =
                *(const uint4*)(Vh + batOff + (size_t)(kb + row) * EMB + headOff + col);
            const unsigned short* vs = (const unsigned short*)&vq;
#pragma unroll
            for (int j = 0; j < 8; j++) Vt[col + j][row] = vs[j];
        }
        wait_async_all();
        __syncthreads();

        // S = Q @ K^T  (gemm-K = Hd = 64 -> two K=32 WMMA steps)
        v8f s[2][4];
#pragma unroll
        for (int mi = 0; mi < 2; mi++)
#pragma unroll
            for (int ni = 0; ni < 4; ni++) s[mi][ni] = vzero;

#pragma unroll
        for (int ks = 0; ks < 64; ks += 32) {
            Frag qa[2], kf[4];
#pragma unroll
            for (int mi = 0; mi < 2; mi++) {
                const unsigned short* p = &Qs[wid * 32 + mi * 16 + lm][ks];
                qa[mi].q[0] = *(const uint4*)(p + 8 * g);
                qa[mi].q[1] = *(const uint4*)(p + 16 + 8 * g);
            }
#pragma unroll
            for (int ni = 0; ni < 4; ni++) {
                const unsigned short* p = &Ks[ni * 16 + lm][ks];
                kf[ni].q[0] = *(const uint4*)(p + 8 * g);
                kf[ni].q[1] = *(const uint4*)(p + 16 + 8 * g);
            }
#pragma unroll
            for (int mi = 0; mi < 2; mi++)
#pragma unroll
                for (int ni = 0; ni < 4; ni++)
                    s[mi][ni] = __builtin_amdgcn_wmma_f32_16x16x32_bf16(
                        false, qa[mi].v, false, kf[ni].v,
                        (short)0, s[mi][ni], false, false);
        }

        // Mask (before scale, like reference), scale, online softmax update.
#pragma unroll
        for (int mi = 0; mi < 2; mi++) {
#pragma unroll
            for (int r = 0; r < 8; r++) {
                const int qi = q0 + wid * 32 + mi * 16 + r + 8 * g;
                float rowm = NEG_INF;
#pragma unroll
                for (int ni = 0; ni < 4; ni++) {
                    int ki  = kb + ni * 16 + lm;
                    float v = s[mi][ni][r];
                    v = (ki > qi) ? NEG_INF : v * 0.125f;  // 1/sqrt(64)
                    s[mi][ni][r] = v;
                    rowm = fmaxf(rowm, v);
                }
                rowm = fmaxf(rowm, __shfl_xor(rowm, 1));
                rowm = fmaxf(rowm, __shfl_xor(rowm, 2));
                rowm = fmaxf(rowm, __shfl_xor(rowm, 4));
                rowm = fmaxf(rowm, __shfl_xor(rowm, 8));

                float mnew  = fmaxf(mrun[mi][r], rowm);
                float mcl   = (mnew == NEG_INF) ? 0.0f : mnew;
                float alpha = exp2f((mrun[mi][r] - mcl) * LOG2E);

                float rsum = 0.0f;
#pragma unroll
                for (int ni = 0; ni < 4; ni++) {
                    float p = exp2f((s[mi][ni][r] - mcl) * LOG2E);
                    rsum += p;
                    Ps[wid * 32 + mi * 16 + r + 8 * g][ni * 16 + lm] =
                        f32_to_bf16_rn(p);
                }
                rsum += __shfl_xor(rsum, 1);
                rsum += __shfl_xor(rsum, 2);
                rsum += __shfl_xor(rsum, 4);
                rsum += __shfl_xor(rsum, 8);

                lrun[mi][r] = lrun[mi][r] * alpha + rsum;
                mrun[mi][r] = mnew;
#pragma unroll
                for (int ni = 0; ni < 4; ni++) o[mi][ni][r] *= alpha;
            }
        }

        // O += P @ V  (gemm-K = 64 keys; Vt gives K-contiguous B rows)
#pragma unroll
        for (int ks = 0; ks < 64; ks += 32) {
            Frag pa[2], vb[4];
#pragma unroll
            for (int mi = 0; mi < 2; mi++) {
                const unsigned short* p = &Ps[wid * 32 + mi * 16 + lm][ks];
                pa[mi].q[0] = *(const uint4*)(p + 8 * g);
                pa[mi].q[1] = *(const uint4*)(p + 16 + 8 * g);
            }
#pragma unroll
            for (int ni = 0; ni < 4; ni++) {
                const unsigned short* p = &Vt[ni * 16 + lm][ks];
                vb[ni].q[0] = *(const uint4*)(p + 8 * g);
                vb[ni].q[1] = *(const uint4*)(p + 16 + 8 * g);
            }
#pragma unroll
            for (int mi = 0; mi < 2; mi++)
#pragma unroll
                for (int ni = 0; ni < 4; ni++)
                    o[mi][ni] = __builtin_amdgcn_wmma_f32_16x16x32_bf16(
                        false, pa[mi].v, false, vb[ni].v,
                        (short)0, o[mi][ni], false, false);
        }
        __syncthreads();
    }

    // Normalize and write context (bf16, heads interleaved back into D=1024).
#pragma unroll
    for (int mi = 0; mi < 2; mi++)
#pragma unroll
        for (int r = 0; r < 8; r++) {
            float l   = lrun[mi][r];
            float inv = (l > 0.0f) ? (1.0f / l) : 0.0f;
            int t = q0 + wid * 32 + mi * 16 + r + 8 * g;
#pragma unroll
            for (int ni = 0; ni < 4; ni++) {
                Ctx[batOff + (size_t)t * EMB + headOff + ni * 16 + lm] =
                    f32_to_bf16_rn(o[mi][ni][r] * inv);
            }
        }
}

// ---------------------------------------------------------------------------
// Host launcher
// ---------------------------------------------------------------------------
extern "C" void kernel_launch(void* const* d_in, const int* in_sizes, int n_in,
                              void* d_out, int out_size, void* d_ws, size_t ws_size,
                              hipStream_t stream) {
    (void)in_sizes; (void)n_in; (void)out_size; (void)ws_size;

    const float* X  = (const float*)d_in[0];
    const float* Wq = (const float*)d_in[1];
    const float* Wk = (const float*)d_in[2];
    const float* Wv = (const float*)d_in[3];
    const float* Wo = (const float*)d_in[4];
    const float* bo = (const float*)d_in[5];
    float* out = (float*)d_out;

    char* ws = (char*)d_ws;
    unsigned short* Xh  = (unsigned short*)(ws + ((size_t)0  << 20));  // 8 MiB
    unsigned short* Wqh = (unsigned short*)(ws + ((size_t)8  << 20));  // 2 MiB
    unsigned short* Wkh = (unsigned short*)(ws + ((size_t)10 << 20));
    unsigned short* Wvh = (unsigned short*)(ws + ((size_t)12 << 20));
    unsigned short* Woh = (unsigned short*)(ws + ((size_t)14 << 20));
    unsigned short* Qh  = (unsigned short*)(ws + ((size_t)16 << 20));  // 8 MiB
    unsigned short* Kh  = (unsigned short*)(ws + ((size_t)24 << 20));
    unsigned short* Vh  = (unsigned short*)(ws + ((size_t)32 << 20));
    unsigned short* Ch  = (unsigned short*)(ws + ((size_t)40 << 20));  // end 48 MiB

    // fp32 -> bf16 converts
    {
        int n4 = MTOT * EMB / 4;      // 1,048,576
        cvt_f32_bf16<<<n4 / 256, 256, 0, stream>>>((const float4*)X, (ushort4*)Xh, n4);
        int w4 = EMB * EMB / 4;       // 262,144
        cvt_f32_bf16<<<w4 / 256, 256, 0, stream>>>((const float4*)Wq, (ushort4*)Wqh, w4);
        cvt_f32_bf16<<<w4 / 256, 256, 0, stream>>>((const float4*)Wk, (ushort4*)Wkh, w4);
        cvt_f32_bf16<<<w4 / 256, 256, 0, stream>>>((const float4*)Wv, (ushort4*)Wvh, w4);
        cvt_f32_bf16<<<w4 / 256, 256, 0, stream>>>((const float4*)Wo, (ushort4*)Woh, w4);
    }

    dim3 gg(EMB / GT_N, MTOT / GT_M);   // (8, 32)
    gemm_bf16_nt<<<gg, 256, 0, stream>>>(Xh, Wqh, Qh, nullptr, nullptr, MTOT, EMB, EMB);
    gemm_bf16_nt<<<gg, 256, 0, stream>>>(Xh, Wkh, Kh, nullptr, nullptr, MTOT, EMB, EMB);
    gemm_bf16_nt<<<gg, 256, 0, stream>>>(Xh, Wvh, Vh, nullptr, nullptr, MTOT, EMB, EMB);

    dim3 ag(SEQ / AQ, NHEADS, BATCH);   // (16, 16, 2)
    attn_causal<<<ag, 128, 0, stream>>>(Qh, Kh, Vh, Ch);

    gemm_bf16_nt<<<gg, 256, 0, stream>>>(Ch, Woh, nullptr, out, bo, MTOT, EMB, EMB);
}
